// GraphNetv2_70927089926261
// MI455X (gfx1250) — matrix operand order repaired and checked by reading, not compiled, not run on hardware
//
#include <hip/hip_runtime.h>

typedef __attribute__((ext_vector_type(16))) __bf16 v16bf;
typedef __attribute__((ext_vector_type(8)))  float  v8f;
typedef unsigned short ushort_t;

#define Bb 256
#define Pp 48
#define Nn 64
#define DeC 64
#define Hc 128
#define DoC 64
#define Tt 5
#define NRe 4032            // N*(N-1); note 4032 == 64*63
#define ECOLS (Bb*NRe)      // 1032192
#define NCOLS (Bb*Nn)       // 16384

union Frag { v16bf v; uint4 q[2]; };

// native hardware bf16 conversion (RNE) -> raw bits
__device__ __forceinline__ ushort_t f2bfbits(float f) {
    union { __bf16 h; ushort_t u; } cv;
    cv.h = (__bf16)f;
    return cv.u;
}

// A fragment: 16x32 bf16 tile from row-major LDS [row][k], stride in ushorts.
// ISA layout: lane m=lane&15 holds row m; K = hl*8+0..7 and 16+hl*8+0..7.
__device__ __forceinline__ v16bf load_afrag(const ushort_t* base, int stride,
                                            int row0, int k0, int lane) {
    int m = lane & 15, hl = lane >> 4;
    const ushort_t* p = base + (row0 + m) * stride + k0 + hl * 8;
    Frag u;
    u.q[0] = *(const uint4*)p;
    u.q[1] = *(const uint4*)(p + 16);
    return u.v;
}

// B fragment: 32x16 (KxN) from row-major weights [n][k] (O x I), stride=I.
// ISA layout: lane holds column n=n0+(lane&15), K = hl*8+0..7 and 16+hl*8+0..7 (hl=lane>>4).
__device__ __forceinline__ v16bf load_bfrag(const ushort_t* W, int istride,
                                            int n0, int k0, int lane) {
    int n = n0 + (lane & 15), kh = lane >> 4;
    const ushort_t* p = W + n * istride + k0 + kh * 8;
    Frag u;
    u.q[0] = *(const uint4*)p;
    u.q[1] = *(const uint4*)(p + 16);
    return u.v;
}

// D fragment (16x16 f32): lane holds N = n0+(lane&15), rows M = v + 8*(lane>>4).
// bias comes from LDS (ds_load_b32), relu is v_max_num_f32, cvt is v_cvt_pk_bf16_f32.
__device__ __forceinline__ void store_h_relu(ushort_t* Hb, int stride, int row0,
                                             int n0, int lane, v8f c, const float* biasLDS) {
    int n = n0 + (lane & 15), hl = lane >> 4;
    float bv = biasLDS[n];
#pragma unroll
    for (int v = 0; v < 8; ++v) {
        float x = fmaxf(c[v] + bv, 0.f);
        Hb[(row0 + v + 8 * hl) * stride + n] = f2bfbits(x);
    }
}

// ---------------- small prep kernels ----------------

__global__ void zero_f32(float* p, int n) {
    int i = blockIdx.x * blockDim.x + threadIdx.x;
    if (i < n) p[i] = 0.f;
}

__global__ void bn_stats(const float* __restrict__ x, const float* __restrict__ gamma,
                         const float* __restrict__ beta, float* scale, float* shift) {
    int p = blockIdx.x;                 // 0..47
    __shared__ float s1[256], s2[256];
    float sum = 0.f, sq = 0.f;
    for (int i = threadIdx.x; i < Bb * Nn; i += 256) {
        int b = i >> 6, n = i & 63;
        float v = x[(b * Pp + p) * Nn + n];
        sum += v; sq += v * v;
    }
    s1[threadIdx.x] = sum; s2[threadIdx.x] = sq;
    __syncthreads();
    for (int o = 128; o > 0; o >>= 1) {
        if (threadIdx.x < o) {
            s1[threadIdx.x] += s1[threadIdx.x + o];
            s2[threadIdx.x] += s2[threadIdx.x + o];
        }
        __syncthreads();
    }
    if (threadIdx.x == 0) {
        float m   = s1[0] * (1.f / (Bb * Nn));
        float var = s2[0] * (1.f / (Bb * Nn)) - m * m;
        float inv = rsqrtf(var + 1e-5f);
        float sc  = gamma[p] * inv;
        scale[p] = sc;
        shift[p] = beta[p] - m * sc;
    }
}

// xn (normalized) -> bf16, transposed to (B, N, P) for contiguous node vectors
__global__ void make_xnT(const float* __restrict__ x, const float* __restrict__ scale,
                         const float* __restrict__ shift, ushort_t* __restrict__ xnT) {
    int t = blockIdx.x * blockDim.x + threadIdx.x;
    if (t >= Bb * Pp * Nn) return;
    int b = t / (Pp * Nn);
    int rem = t - b * (Pp * Nn);
    int p = rem >> 6, n = rem & 63;
    float v = x[t] * scale[p] + shift[p];
    xnT[(b * Nn + n) * Pp + p] = f2bfbits(v);
}

__global__ void cvt_bf16(const float* __restrict__ src, ushort_t* __restrict__ dst, int n) {
    int i = blockIdx.x * blockDim.x + threadIdx.x;
    if (i < n) dst[i] = f2bfbits(src[i]);
}

// fo1 (256 x 112) -> bf16 padded to (256 x 128)
__global__ void cvt_fo1_pad(const float* __restrict__ src, ushort_t* __restrict__ dst) {
    int i = blockIdx.x * blockDim.x + threadIdx.x;   // 0..32767
    int o = i >> 7, k = i & 127;
    dst[i] = (k < 112) ? f2bfbits(src[o * 112 + k]) : (ushort_t)0;
}

// ---------------- edge MLP: (B*NR) x [96 ->256 ->128 ->64], scatter-add to Ebar ----------------
// LDS (ushorts): W1 24576 | W2 32768 | W3 8192 | A 128x96 | H1 128x256 | H2 128x128
//                | Esl 4x64 f32 | bias 448 f32
__global__ __launch_bounds__(256) void edge_mlp(
    const ushort_t* __restrict__ xnT, const int* __restrict__ recv,
    const int* __restrict__ send, const ushort_t* __restrict__ EW,
    const float* __restrict__ b1, const float* __restrict__ b2,
    const float* __restrict__ b3, float* __restrict__ Ebar) {
    extern __shared__ char smem[];
    ushort_t* W1 = (ushort_t*)smem;
    ushort_t* W2 = W1 + 256 * 96;
    ushort_t* W3 = W2 + 128 * 256;
    ushort_t* Ab = W3 + 64 * 128;
    ushort_t* H1 = Ab + 128 * 96;
    ushort_t* H2 = H1 + 128 * 256;
    float*    Esl = (float*)(H2 + 128 * 128);        // 4 receiver slots x 64 feats
    float*    Bl1 = Esl + 256;                       // 256 biases
    float*    Bl2 = Bl1 + 256;                       // 128 biases
    float*    Bl3 = Bl2 + 128;                       // 64 biases

    int tid = threadIdx.x, lane = tid & 31, wave = tid >> 5;
    int col0 = blockIdx.x * 128;
    int g0 = col0 / 63;                              // global receiver-group of first column

    __builtin_prefetch(EW, 0, 0);

    // stage all edge weights (131072 B) into LDS
    {
        const uint4* s = (const uint4*)EW;
        uint4* d = (uint4*)smem;
        for (int i = tid; i < (256 * 96 + 128 * 256 + 64 * 128) / 8; i += 256) d[i] = s[i];
    }
    // stage biases into LDS (remove per-tile global loads from the hot loop)
    Bl1[tid] = b1[tid];
    if (tid < 128) Bl2[tid] = b2[tid];
    if (tid < 64)  Bl3[tid] = b3[tid];
    // stage A tile: 128 columns x 96 (recv feats | send feats)
    {
        int cl = tid >> 1, hl = tid & 1;
        int c = col0 + cl;
        int b = c / NRe, e = c - b * NRe;
        int node = hl ? send[e] : recv[e];
        const uint4* s = (const uint4*)(xnT + (b * Nn + node) * Pp);
        uint4* d = (uint4*)(Ab + cl * 96 + hl * 48);
#pragma unroll
        for (int i = 0; i < 6; ++i) d[i] = s[i];
    }
    Esl[tid] = 0.f;                                  // 256 == 4*64 accumulator slots
    __syncthreads();

    int r0 = wave * 16;

    // layer 1: K=96 (3 chunks), O=256 (16 tiles)
    v16bf a1[3];
#pragma unroll
    for (int k = 0; k < 3; ++k) a1[k] = load_afrag(Ab, 96, r0, k * 32, lane);
#pragma unroll
    for (int nt = 0; nt < 16; ++nt) {
        v8f c = {};
#pragma unroll
        for (int k = 0; k < 3; ++k) {
            v16bf bf = load_bfrag(W1, 96, nt * 16, k * 32, lane);
            c = __builtin_amdgcn_wmma_f32_16x16x32_bf16(false, a1[k], false, bf,
                                                        (short)0, c, false, false);
        }
        store_h_relu(H1, 256, r0, nt * 16, lane, c, Bl1);
    }
    __syncthreads();

    // layer 2: K=256 (8 chunks), O=128 (8 tiles)
    v16bf a2[8];
#pragma unroll
    for (int k = 0; k < 8; ++k) a2[k] = load_afrag(H1, 256, r0, k * 32, lane);
#pragma unroll
    for (int nt = 0; nt < 8; ++nt) {
        v8f c = {};
#pragma unroll
        for (int k = 0; k < 8; ++k) {
            v16bf bf = load_bfrag(W2, 256, nt * 16, k * 32, lane);
            c = __builtin_amdgcn_wmma_f32_16x16x32_bf16(false, a2[k], false, bf,
                                                        (short)0, c, false, false);
        }
        store_h_relu(H2, 128, r0, nt * 16, lane, c, Bl2);
    }
    __syncthreads();

    // layer 3: K=128 (4 chunks), O=64 (4 tiles) -> accumulate per-receiver in LDS
    v16bf a3[4];
#pragma unroll
    for (int k = 0; k < 4; ++k) a3[k] = load_afrag(H2, 128, r0, k * 32, lane);

    int hl = lane >> 4;
    int slot_v[8];
#pragma unroll
    for (int v = 0; v < 8; ++v) {
        int c = col0 + r0 + v + 8 * hl;     // global edge column of this D row
        slot_v[v] = c / 63 - g0;            // receiver-group slot within block (0..3)
    }
#pragma unroll
    for (int nt = 0; nt < 4; ++nt) {
        v8f c = {};
#pragma unroll
        for (int k = 0; k < 4; ++k) {
            v16bf bf = load_bfrag(W3, 128, nt * 16, k * 32, lane);
            c = __builtin_amdgcn_wmma_f32_16x16x32_bf16(false, a3[k], false, bf,
                                                        (short)0, c, false, false);
        }
        int d = nt * 16 + (lane & 15);
        float bv = Bl3[d];
#pragma unroll
        for (int v = 0; v < 8; ++v) {
            float x = fmaxf(c[v] + bv, 0.f);
            atomicAdd(&Esl[slot_v[v] * 64 + d], x);   // ds_add_f32
        }
    }
    __syncthreads();

    // flush: one (slot, feature) per thread -> at most 256 global atomics per block
    {
        int slot = tid >> 6, d = tid & 63;
        int g = g0 + slot;                            // global receiver-group id
        int gmax = (col0 + 127) / 63;
        if (g <= gmax) {
            int b = g >> 6;
            int egrp = (g & 63) * 63;                 // first edge of this group
            int r = recv[egrp];
            atomicAdd(&Ebar[(b * Nn + r) * DeC + d], Esl[slot * 64 + d]);
        }
    }
}

// ---------------- node MLP: (B*N) x [112(pad128) ->256 ->128 ->64], reduce to Nsum ----------------
// LDS (ushorts): W1 32768 | W2 32768 | W3 8192 | A 128x128 | H1 128x256 | H2 128x128 | bias 448 f32
__global__ __launch_bounds__(256) void node_mlp(
    const ushort_t* __restrict__ xnT, const float* __restrict__ Ebar,
    const ushort_t* __restrict__ NW, const float* __restrict__ b1,
    const float* __restrict__ b2, const float* __restrict__ b3,
    float* __restrict__ Nsum) {
    extern __shared__ char smem[];
    ushort_t* W1 = (ushort_t*)smem;
    ushort_t* W2 = W1 + 256 * 128;
    ushort_t* W3 = W2 + 128 * 256;
    ushort_t* Ab = W3 + 64 * 128;
    ushort_t* H1 = Ab + 128 * 128;
    ushort_t* H2 = H1 + 128 * 256;
    float*    Bl1 = (float*)(H2 + 128 * 128);        // 256 biases
    float*    Bl2 = Bl1 + 256;                       // 128 biases
    float*    Bl3 = Bl2 + 128;                       // 64 biases

    int tid = threadIdx.x, lane = tid & 31, wave = tid >> 5;
    int col0 = blockIdx.x * 128;

    {
        const uint4* s = (const uint4*)NW;
        uint4* d = (uint4*)smem;
        for (int i = tid; i < (256 * 128 + 128 * 256 + 64 * 128) / 8; i += 256) d[i] = s[i];
    }
    Bl1[tid] = b1[tid];
    if (tid < 128) Bl2[tid] = b2[tid];
    if (tid < 64)  Bl3[tid] = b3[tid];
    // stage A: col c -> [xnT(48) | Ebar(64) | zeros(16)]
    {
        int cl = tid >> 1, hl = tid & 1;
        int c = col0 + cl;
        int b = c >> 6, n = c & 63;
        ushort_t* Arow = Ab + cl * 128;
        if (hl == 0) {
            const uint4* s = (const uint4*)(xnT + (b * Nn + n) * Pp);
            uint4* d = (uint4*)Arow;
#pragma unroll
            for (int i = 0; i < 6; ++i) d[i] = s[i];
            uint4 z = {0, 0, 0, 0};
            ((uint4*)(Arow + 112))[0] = z;
            ((uint4*)(Arow + 112))[1] = z;
        } else {
            const float2* s = (const float2*)(Ebar + (b * Nn + n) * DeC);
            unsigned* d = (unsigned*)(Arow + 48);
#pragma unroll
            for (int j = 0; j < 32; ++j) {
                float2 v2 = s[j];
                unsigned lo = f2bfbits(v2.x), hi = f2bfbits(v2.y);
                d[j] = (hi << 16) | lo;               // packed bf16 pair
            }
        }
    }
    __syncthreads();

    int r0 = wave * 16;
    int bw = (col0 + r0) >> 6;   // batch of this wave's 16 columns (16 | 64)

    v16bf a1[4];
#pragma unroll
    for (int k = 0; k < 4; ++k) a1[k] = load_afrag(Ab, 128, r0, k * 32, lane);
#pragma unroll
    for (int nt = 0; nt < 16; ++nt) {
        v8f c = {};
#pragma unroll
        for (int k = 0; k < 4; ++k) {
            v16bf bf = load_bfrag(W1, 128, nt * 16, k * 32, lane);
            c = __builtin_amdgcn_wmma_f32_16x16x32_bf16(false, a1[k], false, bf,
                                                        (short)0, c, false, false);
        }
        store_h_relu(H1, 256, r0, nt * 16, lane, c, Bl1);
    }
    __syncthreads();

    v16bf a2[8];
#pragma unroll
    for (int k = 0; k < 8; ++k) a2[k] = load_afrag(H1, 256, r0, k * 32, lane);
#pragma unroll
    for (int nt = 0; nt < 8; ++nt) {
        v8f c = {};
#pragma unroll
        for (int k = 0; k < 8; ++k) {
            v16bf bf = load_bfrag(W2, 256, nt * 16, k * 32, lane);
            c = __builtin_amdgcn_wmma_f32_16x16x32_bf16(false, a2[k], false, bf,
                                                        (short)0, c, false, false);
        }
        store_h_relu(H2, 128, r0, nt * 16, lane, c, Bl2);
    }
    __syncthreads();

    v16bf a3[4];
#pragma unroll
    for (int k = 0; k < 4; ++k) a3[k] = load_afrag(H2, 128, r0, k * 32, lane);
#pragma unroll
    for (int nt = 0; nt < 4; ++nt) {
        v8f c = {};
#pragma unroll
        for (int k = 0; k < 4; ++k) {
            v16bf bf = load_bfrag(W3, 128, nt * 16, k * 32, lane);
            c = __builtin_amdgcn_wmma_f32_16x16x32_bf16(false, a3[k], false, bf,
                                                        (short)0, c, false, false);
        }
        int d = nt * 16 + (lane & 15);
        float bv = Bl3[d];
        float acc = 0.f;
#pragma unroll
        for (int v = 0; v < 8; ++v) acc += fmaxf(c[v] + bv, 0.f);
        acc += __shfl_xor(acc, 16, 32);        // merge the two M-halves of D
        if (lane < 16) atomicAdd(&Nsum[bw * DoC + d], acc);
    }
}

// ---------------- tiny fc chain: (256,64) -> 25 -> 15 -> 5 (no relu) ----------------
__global__ void fc_chain(const float* __restrict__ Nsum,
                         const float* __restrict__ w1, const float* __restrict__ b1,
                         const float* __restrict__ w2, const float* __restrict__ b2,
                         const float* __restrict__ w3, const float* __restrict__ b3,
                         float* __restrict__ out) {
    int b = threadIdx.x;                       // one batch row per thread, 256 threads
    const float* xs = Nsum + b * 64;
    float h1[25], h2[15];
#pragma unroll
    for (int o = 0; o < 25; ++o) {
        float s = b1[o];
        for (int i = 0; i < 64; ++i) s += w1[o * 64 + i] * xs[i];
        h1[o] = s;
    }
#pragma unroll
    for (int o = 0; o < 15; ++o) {
        float s = b2[o];
        for (int i = 0; i < 25; ++i) s += w2[o * 25 + i] * h1[i];
        h2[o] = s;
    }
#pragma unroll
    for (int o = 0; o < 5; ++o) {
        float s = b3[o];
        for (int i = 0; i < 15; ++i) s += w3[o * 15 + i] * h2[i];
        out[b * 5 + o] = s;
    }
}

extern "C" void kernel_launch(void* const* d_in, const int* in_sizes, int n_in,
                              void* d_out, int out_size, void* d_ws, size_t ws_size,
                              hipStream_t stream) {
    const float* x     = (const float*)d_in[0];
    const int*   recv  = (const int*)d_in[1];
    const int*   send  = (const int*)d_in[2];
    const float* gamma = (const float*)d_in[3];
    const float* beta  = (const float*)d_in[4];
    const float* fr1_w = (const float*)d_in[5];
    const float* fr1_b = (const float*)d_in[6];
    const float* fr2_w = (const float*)d_in[7];
    const float* fr2_b = (const float*)d_in[8];
    const float* fr3_w = (const float*)d_in[9];
    const float* fr3_b = (const float*)d_in[10];
    const float* fo1_w = (const float*)d_in[11];
    const float* fo1_b = (const float*)d_in[12];
    const float* fo2_w = (const float*)d_in[13];
    const float* fo2_b = (const float*)d_in[14];
    const float* fo3_w = (const float*)d_in[15];
    const float* fo3_b = (const float*)d_in[16];
    const float* fc1_w = (const float*)d_in[17];
    const float* fc1_b = (const float*)d_in[18];
    const float* fc2_w = (const float*)d_in[19];
    const float* fc2_b = (const float*)d_in[20];
    const float* fc3_w = (const float*)d_in[21];
    const float* fc3_b = (const float*)d_in[22];
    float* out = (float*)d_out;

    // workspace layout (byte offsets)
    char* ws = (char*)d_ws;
    float*    scale = (float*)(ws + 0);
    float*    shift = (float*)(ws + 256);
    ushort_t* xnT   = (ushort_t*)(ws + 512);                 // 786432 ushorts
    ushort_t* EW    = (ushort_t*)(ws + 1573376);             // 65536 ushorts
    ushort_t* NW    = (ushort_t*)(ws + 1704448);             // 73728 ushorts
    float*    Ebar  = (float*)(ws + 1851904);                // 1048576 f32
    float*    Nsum  = (float*)(ws + 6046208);                // 16384 f32 (contiguous after Ebar)

    // 1) zero Ebar + Nsum (contiguous)
    zero_f32<<<(1048576 + 16384 + 255) / 256, 256, 0, stream>>>(Ebar, 1048576 + 16384);

    // 2) BN stats -> folded scale/shift
    bn_stats<<<Pp, 256, 0, stream>>>(x, gamma, beta, scale, shift);

    // 3) normalized, transposed bf16 activations
    make_xnT<<<(Bb * Pp * Nn + 255) / 256, 256, 0, stream>>>(x, scale, shift, xnT);

    // 4) weight conversion to bf16 (fo1 padded K 112->128)
    cvt_bf16<<<(24576 + 255) / 256, 256, 0, stream>>>(fr1_w, EW, 24576);
    cvt_bf16<<<(32768 + 255) / 256, 256, 0, stream>>>(fr2_w, EW + 24576, 32768);
    cvt_bf16<<<(8192 + 255) / 256, 256, 0, stream>>>(fr3_w, EW + 57344, 8192);
    cvt_fo1_pad<<<32768 / 256, 256, 0, stream>>>(fo1_w, NW);
    cvt_bf16<<<(32768 + 255) / 256, 256, 0, stream>>>(fo2_w, NW + 32768, 32768);
    cvt_bf16<<<(8192 + 255) / 256, 256, 0, stream>>>(fo3_w, NW + 65536, 8192);

    // 5) edge MLP (bulk of the FLOPs): 8064 blocks, 256768 B dynamic LDS
    edge_mlp<<<ECOLS / 128, 256, 256768, stream>>>(xnT, recv, send, EW,
                                                   fr1_b, fr2_b, fr3_b, Ebar);

    // 6) node MLP: 128 blocks, 280320 B dynamic LDS
    node_mlp<<<NCOLS / 128, 256, 280320, stream>>>(xnT, Ebar, NW,
                                                   fo1_b, fo2_b, fo3_b, Nsum);

    // 7) tiny fc chain
    fc_chain<<<1, 256, 0, stream>>>(Nsum, fc1_w, fc1_b, fc2_w, fc2_b, fc3_w, fc3_b, out);
}
